// AliasFreeActivation_20486994002477
// MI455X (gfx1250) — matrix operand order repaired and checked by reading, not compiled, not run on hardware
//
#include <hip/hip_runtime.h>

// ---------------------------------------------------------------------------
// AliasFreeActivation for MI455X (gfx1250, wave32, WMMA)
//   x: (8,64,128,128) f32  -> bilinear up x2 -> leaky relu
//   -> depthwise 31x31 conv (single shared sinc filter) -> bilinear down -> f32
//
// The 31x31 depthwise conv (~64.5 GFLOP, ~1000 FLOP/byte) runs on
// v_wmma_f32_16x16x32_f16 via a Toeplitz formulation along x:
//   out[y, x+n] = sum_dy sum_j act[y+dy, x+j] * T_dy[j, n],  T_dy[j,n]=w[dy][j-n]
// M=16 output rows, N=16 output cols, K=64 (two K=32 WMMAs), accumulated over
// the 31 filter rows dy. Each wave computes TWO adjacent x-tiles so the
// B fragments (b0, b1) are shared by both tiles and
// amortized: 6 LDS loads -> 4 WMMAs per dy (vs 8 -> 2 before).
// ---------------------------------------------------------------------------

typedef __attribute__((ext_vector_type(16))) _Float16 v16h;
typedef __attribute__((ext_vector_type(8)))  _Float16 v8h;
typedef __attribute__((ext_vector_type(8)))  float    v8f;

#define NIMG      512            // 8*64 independent channel images
#define H_UP      256
#define W_UP      256
#define HP        288            // padded rows (need 286: 256 + 30)
#define WP        304            // padded cols (need up to x+63 = 303)
#define ACT_ELEMS ((size_t)NIMG * HP * WP)                 // f16
#define ACT_BYTES (ACT_ELEMS * 2)                          // 89,653,248
#define FILT_BYTES ((size_t)NIMG * H_UP * W_UP * 4)        // 134,217,728
#define TOE_OFF   (ACT_BYTES + FILT_BYTES)                 // 223,870,976 (32B aligned)

// Toeplitz table layout (f16 halves), matching per-lane v16h LDS reads:
//   chunk0: [0, 15872)  = dy*512 + lane*16 + h   (all 32 lanes, K=0..31)
//   chunk1: [15872, 23808) = dy*256 + lane*16 + h (lanes 0..15 only, K=32..47)
//   zeros : [23808, 23824) shared 32B zero slot for lanes 16..31 of chunk1
#define TOE_HALVES 23824

// conv tiling
#define SA_COLS   176            // 128 output cols + 48 halo (stride: 352 B, 16B-aligned)
#define SA_ROWS   46             // 16 output rows + 30 halo

// -------------------------- kernel 0: Toeplitz build ------------------------
__global__ void build_toeplitz(const float* __restrict__ filt,
                               _Float16* __restrict__ toe) {
    int idx = blockIdx.x * 256 + threadIdx.x;
    if (idx >= TOE_HALVES) return;
    float val = 0.0f;
    if (idx < 15872) {                       // chunk0: j = (lane/16)*16 + h
        int dy = idx >> 9, rem = idx & 511;
        int lane = rem >> 4, h = rem & 15;
        int n = lane & 15;
        int j = ((lane >> 4) << 4) + h;
        int d = j - n;
        if (d >= 0 && d < 31) val = filt[dy * 31 + d];
    } else if (idx < 23808) {                // chunk1 (lanes 0..15): j = 32 + h
        int t = idx - 15872;
        int dy = t >> 8, rem = t & 255;
        int n = rem >> 4, h = rem & 15;
        int d = 32 + h - n;
        if (d >= 0 && d < 31) val = filt[dy * 31 + d];
    }
    toe[idx] = (_Float16)val;
}

// ----------------- kernel 1: bilinear up x2 + leaky relu -> f16 -------------
__global__ void upsample_lrelu(const float* __restrict__ x,
                               _Float16* __restrict__ act, int total) {
    int idx = blockIdx.x * 256 + threadIdx.x;
    if (idx >= total) return;
    int img = idx / (HP * WP);
    int rem = idx % (HP * WP);
    int r = rem / WP, cc = rem % WP;
    int y = r - 15, xx = cc - 15;            // up-image coords (pad 15 baked in)
    float v = 0.0f;
    if ((unsigned)y < 256u && (unsigned)xx < 256u) {
        const float s = 127.0f / 255.0f;     // (in-1)/(out-1), align-corners
        float cy = (float)y * s, cx = (float)xx * s;
        int iy = (int)cy; if (iy > 126) iy = 126;
        int ix = (int)cx; if (ix > 126) ix = 126;
        float wy = cy - (float)iy, wx = cx - (float)ix;
        const float* p = x + (size_t)img * 16384 + iy * 128 + ix;
        float v00 = p[0], v01 = p[1], v10 = p[128], v11 = p[129];
        float top = v00 + (v01 - v00) * wx;
        float bot = v10 + (v11 - v10) * wx;
        v = top + (bot - top) * wy;
        v = v >= 0.0f ? v : 0.01f * v;       // leaky relu
    }
    act[idx] = (_Float16)v;
}

// -------------------- kernel 2: depthwise 31x31 conv (WMMA) -----------------
// Block = 128 threads (4 waves). Each wave computes two adjacent 16x16 output
// tiles; the 4 waves cover 128 consecutive output columns (one xblock).
__global__ __launch_bounds__(128) void conv_wmma(
        const _Float16* __restrict__ act,
        const _Float16* __restrict__ toe,
        float* __restrict__ filtered) {
    __shared__ alignas(32) _Float16 sT[TOE_HALVES];        // 47,648 B
    __shared__ alignas(32) _Float16 sA[SA_ROWS * SA_COLS]; // 16,192 B

    const int tid    = threadIdx.x;
    const int xblock = blockIdx.x;   // 0..1   (128 cols each)
    const int ytile  = blockIdx.y;   // 0..15
    const int img    = blockIdx.z;   // 0..511

    // stage Toeplitz tables (hot in L2)
    {
        const v8h* s = (const v8h*)toe;
        v8h* d = (v8h*)sT;
        for (int i = tid; i < TOE_HALVES / 8; i += 128) d[i] = s[i];
    }
    // stage activation tile: 46 padded rows x 176 cols
    {
        const size_t base = (size_t)img * (HP * WP) + (size_t)(ytile * 16) * WP
                          + (size_t)(xblock * 128);
        for (int i = tid; i < SA_ROWS * (SA_COLS / 8); i += 128) {
            int r  = i / (SA_COLS / 8);
            int cc = (i % (SA_COLS / 8)) * 8;
            *(v8h*)(sA + r * SA_COLS + cc) =
                *(const v8h*)(act + base + (size_t)r * WP + cc);
        }
    }
    __syncthreads();

    const int lane = tid & 31;
    const int wv   = tid >> 5;        // wave -> 32-col slab within block
    const int m    = lane & 15;       // A-matrix row (output y offset)
    const int grp  = lane >> 4;       // lane half -> K sub-range (contiguous K = 16*grp + s)

    // A fragments: lanes 0-15 read K 0-15, lanes 16-31 read K 16-31
    const _Float16* aPtr = sA + m * SA_COLS + wv * 32 + grp * 16;
    // B chunk0: lane l holds N=l&15, K = (l/16)*16 + h
    const _Float16* b0Ptr = sT + lane * 16;
    // B chunk1: lanes 0-15 hold K=32..47; lanes 16-31 (K=48..63) are all-zero
    const _Float16* b1Ptr = sT + (grp == 0 ? 15872 + lane * 16 : 23808);
    const int b1Stride = (grp == 0) ? 256 : 0;

    v8f acc0 = {};
    v8f acc1 = {};
    for (int dy = 0; dy < 31; ++dy) {
        // 4 contiguous 32B A fragments (K = 16*half + s)
        v16h a0t0 = *(const v16h*)(aPtr + 0);
        v16h a0t1 = *(const v16h*)(aPtr + 16);
        v16h a1t0 = *(const v16h*)(aPtr + 32);
        v16h a1t1 = *(const v16h*)(aPtr + 48);
        v16h b0 = *(const v16h*)(b0Ptr + dy * 512);
        v16h b1 = *(const v16h*)(b1Ptr + dy * b1Stride);
        // group by B operand so back-to-back WMMAs share it
        acc0 = __builtin_amdgcn_wmma_f32_16x16x32_f16(false, a0t0, false, b0,
                                                      (short)0, acc0, false, false);
        acc1 = __builtin_amdgcn_wmma_f32_16x16x32_f16(false, a0t1, false, b0,
                                                      (short)0, acc1, false, false);
        acc0 = __builtin_amdgcn_wmma_f32_16x16x32_f16(false, a1t0, false, b1,
                                                      (short)0, acc0, false, false);
        acc1 = __builtin_amdgcn_wmma_f32_16x16x32_f16(false, a1t1, false, b1,
                                                      (short)0, acc1, false, false);
        aPtr += SA_COLS;
    }

    // D layout: VGPR v, lane l -> M = v + 8*(l>=16), N = l&15
    const int xo = xblock * 128 + wv * 32 + (lane & 15);
    const int yb = ytile * 16 + grp * 8;
    float* o = filtered + (size_t)img * (H_UP * W_UP) + (size_t)yb * W_UP + xo;
#pragma unroll
    for (int v = 0; v < 8; ++v) {
        o[v * W_UP]      = acc0[v];
        o[v * W_UP + 16] = acc1[v];
    }
}

// ----------------------- kernel 3: bilinear downsample ----------------------
__global__ void downsample(const float* __restrict__ filtered,
                           float* __restrict__ out, int total) {
    int idx = blockIdx.x * 256 + threadIdx.x;
    if (idx >= total) return;
    int img = idx / 16384;
    int rem = idx % 16384;
    int y = rem / 128, xx = rem % 128;
    const float s = 255.0f / 127.0f;          // (in-1)/(out-1), align-corners
    float cy = (float)y * s, cx = (float)xx * s;
    int iy = (int)cy; if (iy > 254) iy = 254;
    int ix = (int)cx; if (ix > 254) ix = 254;
    float wy = cy - (float)iy, wx = cx - (float)ix;
    const float* p = filtered + (size_t)img * (H_UP * W_UP) + iy * W_UP + ix;
    float v00 = p[0], v01 = p[1], v10 = p[W_UP], v11 = p[W_UP + 1];
    float top = v00 + (v01 - v00) * wx;
    float bot = v10 + (v11 - v10) * wx;
    out[idx] = top + (bot - top) * wy;
}

// ---------------------------------------------------------------------------
extern "C" void kernel_launch(void* const* d_in, const int* in_sizes, int n_in,
                              void* d_out, int out_size, void* d_ws, size_t ws_size,
                              hipStream_t stream) {
    const float* x    = (const float*)d_in[0];   // (8,64,128,128) f32
    const float* filt = (const float*)d_in[1];   // (31,31) f32
    float* out = (float*)d_out;                  // (8,64,128,128) f32

    char* ws = (char*)d_ws;                      // needs ~214 MiB of scratch
    _Float16* act      = (_Float16*)ws;                       // padded f16 activations
    float*    filtered = (float*)(ws + ACT_BYTES);            // f32 conv output
    _Float16* toe      = (_Float16*)(ws + TOE_OFF);           // Toeplitz tables

    build_toeplitz<<<(TOE_HALVES + 255) / 256, 256, 0, stream>>>(filt, toe);

    const int upTotal = (int)(NIMG * HP * WP);                // 44,826,624
    upsample_lrelu<<<(upTotal + 255) / 256, 256, 0, stream>>>(x, act, upTotal);

    dim3 grid(2, 16, NIMG);                                   // xblocks, ytiles, images
    conv_wmma<<<grid, 128, 0, stream>>>(act, toe, filtered);

    const int dnTotal = NIMG * 128 * 128;                     // 8,388,608
    downsample<<<(dnTotal + 255) / 256, 256, 0, stream>>>(filtered, out, dnTotal);
}
